// MambaBlock_5033701670981
// MI455X (gfx1250) — matrix operand order repaired and checked by reading, not compiled, not run on hardware
//
#include <hip/hip_runtime.h>

// ---------------- problem constants ----------------
constexpr int Bn   = 4;
constexpr int Tn   = 8192;
constexpr int Kn   = 64;          // EMB_KS
constexpr int Ln   = Tn - Kn + 1; // 8129
constexpr int LP   = 8144;        // L padded to multiple of 16 (509*16)
constexpr int Mn   = Bn * LP;     // 32576 rows (batch-major, padded)
constexpr int DI   = 256;         // D_INNER
constexpr int DS   = 16;          // D_STATE
constexpr int DTR  = 4;           // DT_RANK
constexpr int XPN  = DTR + 2 * DS;  // 36
constexpr int NCH  = 64;          // scan chunks
constexpr int CHL  = 128;         // chunk length (64*128 >= L)

typedef _Float16 v16h __attribute__((ext_vector_type(16)));
typedef _Float16 v8h  __attribute__((ext_vector_type(8)));
typedef float    v8f  __attribute__((ext_vector_type(8)));

// ---------------- WMMA fragment helpers (CDNA5 16-bit layouts) ----------------
// A (16x32, f16): lanes 0-15 -> M=lane, elems 0..7 = K k0+0..7, 8..15 = K k0+16..23
//                 lanes 16-31 -> M=lane-16, elems 0..7 = K k0+8..15, 8..15 = K k0+24..31
__device__ inline v16h frag_a(const _Float16* p0, int ld, int k0, int lane) {
    int row = lane & 15;
    int kk  = k0 + ((lane & 16) ? 8 : 0);
    const _Float16* p = p0 + row * ld + kk;
    v8h lo = *(const v8h*)(p);
    v8h hi = *(const v8h*)(p + 16);
    v16h f;
#pragma unroll
    for (int i = 0; i < 8; ++i) { f[i] = lo[i]; f[i + 8] = hi[i]; }
    return f;
}
// B (32x16, f16), B[k][n] = W[n][k], W row-major (N x ld):
// lane -> N = n0 + lane%16 ; lanes 0-15 hold K k0+0..7,16..23, lanes 16-31 hold K k0+8..15,24..31
__device__ inline v16h frag_b(const _Float16* w, int ld, int n0, int k0, int lane) {
    int col = n0 + (lane & 15);
    int kk  = k0 + ((lane & 16) ? 8 : 0);
    const _Float16* p = w + col * ld + kk;
    v8h lo = *(const v8h*)(p);
    v8h hi = *(const v8h*)(p + 16);
    v16h f;
#pragma unroll
    for (int i = 0; i < 8; ++i) { f[i] = lo[i]; f[i + 8] = hi[i]; }
    return f;
}
__device__ inline v8f wmma16(v16h a, v16h b, v8f c) {
    return __builtin_amdgcn_wmma_f32_16x16x32_f16(false, a, false, b, (short)0, c, false, false);
}

__device__ inline float silu(float v) { return v / (1.f + __expf(-v)); }

// ---------------- prep: f32->f16 weights, A = -exp(Alog) ----------------
__global__ void k_prep(const float* f_in_w, const float* b_in_w,
                       const float* f_out_w, const float* b_out_w,
                       const float* f_Alog, const float* b_Alog,
                       _Float16* in_w_h, _Float16* out_w_h, float* Aneg) {
    int i = blockIdx.x * blockDim.x + threadIdx.x;
    if (i < 2 * DI * Kn) {            // 512*64
        in_w_h[i]              = (_Float16)f_in_w[i];
        in_w_h[2 * DI * Kn + i] = (_Float16)b_in_w[i];
    }
    if (i < Kn * DI) {                // 64*256
        out_w_h[i]             = (_Float16)f_out_w[i];
        out_w_h[Kn * DI + i]   = (_Float16)b_out_w[i];
    }
    if (i < DI * DS) {                // 256*16
        Aneg[i]            = -__expf(f_Alog[i]);
        Aneg[DI * DS + i]  = -__expf(b_Alog[i]);
    }
}

// ---------------- window extraction + rmsnorm (both directions) ----------------
__global__ __launch_bounds__(256) void k_rms(const float* x, const float* fw, const float* bw,
                                             _Float16* nf, _Float16* nb) {
    int wave = threadIdx.x >> 5, lane = threadIdx.x & 31;
    int row = blockIdx.x * 8 + wave;          // over B*L
    if (row >= Bn * Ln) return;
    int b = row / Ln, l = row % Ln;
    const float* s = x + b * Tn + l;
    float v0 = s[lane], v1 = s[lane + 32];
    float ss = v0 * v0 + v1 * v1;
#pragma unroll
    for (int off = 16; off; off >>= 1) ss += __shfl_xor(ss, off, 32);
    float inv = rsqrtf(ss * (1.f / 64.f) + 1e-5f);
    int mf = b * LP + l;
    int mb = b * LP + (Ln - 1 - l);
    nf[mf * 64 + lane]      = (_Float16)(v0 * inv * fw[lane]);
    nf[mf * 64 + lane + 32] = (_Float16)(v1 * inv * fw[lane + 32]);
    nb[mb * 64 + lane]      = (_Float16)(v0 * inv * bw[lane]);
    nb[mb * 64 + lane + 32] = (_Float16)(v1 * inv * bw[lane + 32]);
}

// ---------------- in-projection GEMM: (M,64) x (64,512) -> xz (M,512), WMMA ----------------
__global__ __launch_bounds__(128) void k_gemm_in(const _Float16* A, const _Float16* W, float* xz) {
    int mtile = blockIdx.x;
    int wave = threadIdx.x >> 5, lane = threadIdx.x & 31;
    const _Float16* Ab = A + mtile * 16 * Kn;
    int nbase = wave * 128;                       // 4 waves x 128 cols
    v8f acc[8];
#pragma unroll
    for (int t = 0; t < 8; ++t)
#pragma unroll
        for (int r = 0; r < 8; ++r) acc[t][r] = 0.f;
#pragma unroll
    for (int k0 = 0; k0 < Kn; k0 += 32) {
        v16h af = frag_a(Ab, Kn, k0, lane);
#pragma unroll
        for (int t = 0; t < 8; ++t) {
            v16h bf = frag_b(W, Kn, nbase + t * 16, k0, lane);
            acc[t] = wmma16(af, bf, acc[t]);
        }
    }
    int mlo = (lane & 16) ? 8 : 0;
    int nc  = lane & 15;
#pragma unroll
    for (int t = 0; t < 8; ++t)
#pragma unroll
        for (int r = 0; r < 8; ++r)
            xz[(mtile * 16 + r + mlo) * 512 + nbase + t * 16 + nc] = acc[t][r];
}

// ---------------- depthwise causal conv(4) + bias + SiLU on x half ----------------
__global__ __launch_bounds__(256) void k_conv_silu(const float* xz, const float* cw,
                                                   const float* cb, float* xc) {
    int idx = blockIdx.x * 256 + threadIdx.x;
    int m = idx >> 8, d = idx & 255;
    int l = m % LP;
    if (l >= Ln) return;
    int b = m / LP;
    float acc = cb[d];
#pragma unroll
    for (int j = 0; j < 4; ++j) {
        int ls = l - 3 + j;
        if (ls >= 0) acc = fmaf(xz[(b * LP + ls) * 512 + d], cw[d * 4 + j], acc);
    }
    xc[m * 256 + d] = silu(acc);
}

// ---------------- x-proj (36 wide) + dt = softplus(dt@dt_w.T + dt_b) ----------------
__global__ __launch_bounds__(256) void k_proj_dt(const float* xc, const float* xpw,
                                                 const float* dtw, const float* dtbv,
                                                 float* xdbl, float* dtb) {
    int m = blockIdx.x;
    int l = m % LP;
    if (l >= Ln) return;
    __shared__ float xr[DI];
    __shared__ float xd[XPN];
    int t = threadIdx.x;
    xr[t] = xc[m * DI + t];
    __syncthreads();
    if (t < XPN) {
        float a = 0.f;
#pragma unroll 8
        for (int k = 0; k < DI; ++k) a = fmaf(xr[k], xpw[t * DI + k], a);
        xd[t] = a;
        xdbl[m * XPN + t] = a;
    }
    __syncthreads();
    float s = dtbv[t];
#pragma unroll
    for (int r = 0; r < DTR; ++r) s = fmaf(xd[r], dtw[t * DTR + r], s);
    dtb[m * DI + t] = (s > 20.f) ? s : log1pf(__expf(s));  // softplus
}

// ---------------- scan pass 1: per-chunk (prod a, partial h) ----------------
__global__ __launch_bounds__(256) void k_scan_part(const float* dtb, const float* xc,
                                                   const float* xdbl, const float* Aneg,
                                                   float* sumA, float* sumH) {
    int ch = blockIdx.x, b = blockIdx.y, d = threadIdx.x;
    int l0 = ch * CHL;
    int l1 = (l0 + CHL < Ln) ? l0 + CHL : Ln;
    float Ad[DS], h[DS], ap[DS];
#pragma unroll
    for (int n = 0; n < DS; ++n) { Ad[n] = Aneg[d * DS + n]; h[n] = 0.f; ap[n] = 1.f; }
    __shared__ float Bs[2][DS];
    for (int l = l0; l < l1; ++l) {
        int m = b * LP + l;
        int buf = l & 1;
        if (d < DS) Bs[buf][d] = xdbl[m * XPN + DTR + d];
        __syncthreads();
        float dtv = dtb[m * DI + d];
        float dx  = dtv * xc[m * DI + d];
#pragma unroll
        for (int n = 0; n < DS; ++n) {
            float a = __expf(dtv * Ad[n]);
            h[n] = fmaf(a, h[n], dx * Bs[buf][n]);
            ap[n] *= a;
        }
    }
    int base = ((b * NCH + ch) * DI + d) * DS;
#pragma unroll
    for (int n = 0; n < DS; ++n) { sumA[base + n] = ap[n]; sumH[base + n] = h[n]; }
}

// ---------------- scan pass 2: sequential chunk fix-up -> per-chunk h0 ----------------
__global__ __launch_bounds__(256) void k_scan_fix(const float* sumA, const float* sumH,
                                                  float* h0c) {
    int i = blockIdx.x * 256 + threadIdx.x;     // (b, d, n) flat: 4*256*16
    if (i >= Bn * DI * DS) return;
    int n = i & 15, d = (i >> 4) & 255, b = i >> 12;
    float h = 0.f;
    for (int ch = 0; ch < NCH; ++ch) {
        int base = ((b * NCH + ch) * DI + d) * DS + n;
        h0c[base] = h;
        h = fmaf(sumA[base], h, sumH[base]);
    }
}

// ---------------- scan pass 3: replay with correct h0, emit gated y (f16) ----------------
__global__ __launch_bounds__(256) void k_scan_emit(const float* dtb, const float* xc,
                                                   const float* xdbl, const float* xz,
                                                   const float* Aneg, const float* Dw,
                                                   const float* h0c, _Float16* yh) {
    int ch = blockIdx.x, b = blockIdx.y, d = threadIdx.x;
    int l0 = ch * CHL;
    int l1 = (l0 + CHL < Ln) ? l0 + CHL : Ln;
    float Ad[DS], h[DS];
    int base = ((b * NCH + ch) * DI + d) * DS;
#pragma unroll
    for (int n = 0; n < DS; ++n) { Ad[n] = Aneg[d * DS + n]; h[n] = h0c[base + n]; }
    float Dd = Dw[d];
    __shared__ float Bs[2][DS], Cs[2][DS];
    for (int l = l0; l < l1; ++l) {
        int m = b * LP + l;
        int buf = l & 1;
        if (d < DS) Bs[buf][d] = xdbl[m * XPN + DTR + d];
        else if (d < 2 * DS) Cs[buf][d - DS] = xdbl[m * XPN + DTR + DS + (d - DS)];
        __syncthreads();
        float dtv = dtb[m * DI + d];
        float xv  = xc[m * DI + d];
        float dx  = dtv * xv;
        float y = 0.f;
#pragma unroll
        for (int n = 0; n < DS; ++n) {
            float a = __expf(dtv * Ad[n]);
            h[n] = fmaf(a, h[n], dx * Bs[buf][n]);
            y = fmaf(h[n], Cs[buf][n], y);
        }
        y = fmaf(xv, Dd, y);
        float z = xz[m * 512 + DI + d];
        yh[m * DI + d] = (_Float16)(y * silu(z));
    }
}

// ---------------- out-projection GEMM (M,256)x(256,64) + residual, WMMA ----------------
__global__ __launch_bounds__(128) void k_gemm_out(const _Float16* Y, const _Float16* Wo,
                                                  const float* sig, int backward, float* res) {
    int mtile = blockIdx.x;
    int wave = threadIdx.x >> 5, lane = threadIdx.x & 31;
    const _Float16* Ab = Y + mtile * 16 * DI;
    int n0 = wave * 16;
    v8f acc;
#pragma unroll
    for (int r = 0; r < 8; ++r) acc[r] = 0.f;
#pragma unroll
    for (int k0 = 0; k0 < DI; k0 += 32) {
        v16h af = frag_a(Ab, DI, k0, lane);
        v16h bf = frag_b(Wo, DI, n0, k0, lane);
        acc = wmma16(af, bf, acc);
    }
    int mlo = (lane & 16) ? 8 : 0;
    int nc  = n0 + (lane & 15);
#pragma unroll
    for (int r = 0; r < 8; ++r) {
        int m = mtile * 16 + r + mlo;
        int b = m / LP, l = m % LP;
        if (l < Ln) {
            int lo = backward ? (Ln - 1 - l) : l;
            res[(b * LP + lo) * Kn + nc] = acc[r] + sig[b * Tn + lo + nc];
        }
    }
}

// ---------------- final 128-ch x 64-tap conv + input residual ----------------
__global__ __launch_bounds__(256) void k_conv_tr(const float* resf, const float* resb,
                                                 const float* ctw, const float* ctb,
                                                 const float* x, float* out) {
    __shared__ float w[128 * 64];
    for (int i = threadIdx.x; i < 128 * 64; i += 256) w[i] = ctw[i];
    __syncthreads();
    int t = blockIdx.x * 256 + threadIdx.x;
    int b = blockIdx.y;
    float acc = ctb[0] + x[b * Tn + t];
    for (int j = 0; j < Kn; ++j) {
        int l = t - j;
        if (l < 0 || l >= Ln) continue;
        const float* rf = resf + (b * LP + l) * Kn;
        const float* rb = resb + (b * LP + l) * Kn;
#pragma unroll 8
        for (int c = 0; c < 64; ++c) acc = fmaf(rf[c], w[c * 64 + j], acc);
#pragma unroll 8
        for (int c = 0; c < 64; ++c) acc = fmaf(rb[c], w[(64 + c) * 64 + j], acc);
    }
    out[b * Tn + t] = acc;
}

// ---------------- orchestration ----------------
extern "C" void kernel_launch(void* const* d_in, const int* in_sizes, int n_in,
                              void* d_out, int out_size, void* d_ws, size_t ws_size,
                              hipStream_t stream) {
    const float* x = (const float*)d_in[0];
    // per-direction parameter tables (setup_inputs order)
    const float* norm_w[2]  = {(const float*)d_in[1],  (const float*)d_in[11]};
    const float* in_w[2]    = {(const float*)d_in[2],  (const float*)d_in[12]};
    const float* conv_w[2]  = {(const float*)d_in[3],  (const float*)d_in[13]};
    const float* conv_b[2]  = {(const float*)d_in[4],  (const float*)d_in[14]};
    const float* xproj_w[2] = {(const float*)d_in[5],  (const float*)d_in[15]};
    const float* dt_w[2]    = {(const float*)d_in[6],  (const float*)d_in[16]};
    const float* dt_b[2]    = {(const float*)d_in[7],  (const float*)d_in[17]};
    const float* Alog[2]    = {(const float*)d_in[8],  (const float*)d_in[18]};
    const float* Dvec[2]    = {(const float*)d_in[9],  (const float*)d_in[19]};
    const float* out_w[2]   = {(const float*)d_in[10], (const float*)d_in[20]};
    const float* ct_w = (const float*)d_in[21];
    const float* ct_b = (const float*)d_in[22];
    float* out = (float*)d_out;

    char* ws = (char*)d_ws;
    size_t off = 0;
    auto alloc = [&](size_t nbytes) -> void* {
        void* p = (void*)(ws + off);
        off += (nbytes + 255) & ~(size_t)255;
        return p;
    };
    _Float16* in_w_h  = (_Float16*)alloc((size_t)2 * 2 * DI * Kn * 2);  // both dirs
    _Float16* out_w_h = (_Float16*)alloc((size_t)2 * Kn * DI * 2);
    float*    Aneg    = (float*)alloc((size_t)2 * DI * DS * 4);
    _Float16* normed  = (_Float16*)alloc((size_t)2 * Mn * Kn * 2);      // f then b
    float*    resf    = (float*)alloc((size_t)Mn * Kn * 4);
    float*    resb    = (float*)alloc((size_t)Mn * Kn * 4);
    // direction-reused scratch
    float*    xz      = (float*)alloc((size_t)Mn * 512 * 4);
    float*    xc      = (float*)alloc((size_t)Mn * DI * 4);
    float*    xdbl    = (float*)alloc((size_t)Mn * XPN * 4);
    float*    dtb     = (float*)alloc((size_t)Mn * DI * 4);
    float*    sumA    = (float*)alloc((size_t)Bn * NCH * DI * DS * 4);
    float*    sumH    = (float*)alloc((size_t)Bn * NCH * DI * DS * 4);
    float*    h0c     = (float*)alloc((size_t)Bn * NCH * DI * DS * 4);
    _Float16* yh      = (_Float16*)alloc((size_t)Mn * DI * 2);
    (void)ws_size; (void)n_in; (void)in_sizes; (void)out_size;

    k_prep<<<128, 256, 0, stream>>>(in_w[0], in_w[1], out_w[0], out_w[1],
                                    Alog[0], Alog[1], in_w_h, out_w_h, Aneg);
    k_rms<<<(Bn * Ln + 7) / 8, 256, 0, stream>>>(x, norm_w[0], norm_w[1],
                                                 normed, normed + (size_t)Mn * Kn);

    for (int dir = 0; dir < 2; ++dir) {
        const _Float16* nrm = normed + (size_t)dir * Mn * Kn;
        const _Float16* wih = in_w_h + (size_t)dir * 2 * DI * Kn;
        const _Float16* woh = out_w_h + (size_t)dir * Kn * DI;
        const float*    An  = Aneg + (size_t)dir * DI * DS;
        float*          res = dir ? resb : resf;

        k_gemm_in<<<Mn / 16, 128, 0, stream>>>(nrm, wih, xz);
        k_conv_silu<<<Mn, 256, 0, stream>>>(xz, conv_w[dir], conv_b[dir], xc);
        k_proj_dt<<<Mn, 256, 0, stream>>>(xc, xproj_w[dir], dt_w[dir], dt_b[dir], xdbl, dtb);
        k_scan_part<<<dim3(NCH, Bn), 256, 0, stream>>>(dtb, xc, xdbl, An, sumA, sumH);
        k_scan_fix<<<(Bn * DI * DS + 255) / 256, 256, 0, stream>>>(sumA, sumH, h0c);
        k_scan_emit<<<dim3(NCH, Bn), 256, 0, stream>>>(dtb, xc, xdbl, xz, An, Dvec[dir], h0c, yh);
        k_gemm_out<<<Mn / 16, 128, 0, stream>>>(yh, woh, x, dir, res);
    }

    k_conv_tr<<<dim3(Tn / 256, Bn), 256, 0, stream>>>(resf, resb, ct_w, ct_b, x, out);
}
